// GoGEncoder_72559177498877
// MI455X (gfx1250) — matrix operand order repaired and checked by reading, not compiled, not run on hardware
//
#include <hip/hip_runtime.h>

// ---------------------------------------------------------------------------
// GCN encoder for MI455X (gfx1250, wave32).
//  - GEMMs via v_wmma_f32_16x16x32_bf16 (bf16 A/B, fp32 acc)
//  - edge aggregation via coalesced gathers + global_atomic_add_f32 (L2-resident)
//  - self-loop term fused into GEMM epilogue, pooling fused into finalize
// ---------------------------------------------------------------------------

typedef __attribute__((ext_vector_type(16))) __bf16 v16bf;
typedef __attribute__((ext_vector_type(8)))  float  v8f;

union Frag256 {            // one WMMA 16-bit operand fragment per lane (32B)
    v16bf v;
    uint4 q[2];
};

static constexpr int NODE_DIM = 47;
static constexpr int HIDDEN   = 128;
static constexpr int LATENT   = 64;
static constexpr int KPAD_EMB = 64;   // pad K=47 -> 64 (two k-steps of 32)

__device__ __forceinline__ unsigned short f2bf(float f) {
    union { float f; unsigned u; } x{f};
    unsigned u = x.u + 0x7FFFu + ((x.u >> 16) & 1u);   // round-to-nearest-even
    return (unsigned short)(u >> 16);
}

// ---------------------------------------------------------------- utilities
__global__ void fill_f32(float* __restrict__ p, float v, int n) {
    int i = blockIdx.x * blockDim.x + threadIdx.x;
    if (i < n) p[i] = v;
}

__global__ void deg_kernel(const int* __restrict__ dst, float* __restrict__ deg, int E) {
    int e = blockIdx.x * blockDim.x + threadIdx.x;
    if (e < E) atomicAdd(&deg[dst[e]], 1.0f);
}

__global__ void rsqrt_kernel(float* __restrict__ d, int n) {
    int i = blockIdx.x * blockDim.x + threadIdx.x;
    if (i < n) d[i] = rsqrtf(d[i]);          // deg -> deg^{-1/2} in place
}

// x[N x 47] fp32 -> xb[N x 64] bf16 (zero padded K)
__global__ void pad_x_bf16(const float* __restrict__ x, unsigned short* __restrict__ xb, int nRows) {
    int idx = blockIdx.x * blockDim.x + threadIdx.x;
    if (idx >= nRows * KPAD_EMB) return;
    int i = idx / KPAD_EMB, k = idx % KPAD_EMB;
    float v = (k < NODE_DIM) ? x[(size_t)i * NODE_DIM + k] : 0.0f;
    xb[idx] = f2bf(v);
}

// Pre-pack W[K x NOUT] fp32 (row-major) into WMMA B-fragment order, bf16.
// Element order: (((tile_n*KS + s)*32 + lane)*16 + e); lane<16: n=t*16+lane,
// k=s*32+e ; lane>=16: n=t*16+lane-16, k=s*32+16+e.
template <int KREAL, int KPAD, int NOUT>
__global__ void pack_w(const float* __restrict__ W, unsigned short* __restrict__ P) {
    constexpr int KS = KPAD / 32;
    constexpr int NT = NOUT / 16;
    int idx = blockIdx.x * blockDim.x + threadIdx.x;
    if (idx >= NT * KS * 32 * 16) return;
    int e    = idx & 15;
    int lane = (idx >> 4) & 31;
    int rest = idx >> 9;                 // t*KS + s
    int s    = rest % KS;
    int t    = rest / KS;
    int n = t * 16 + (lane & 15);
    int k = s * 32 + (lane >> 4) * 16 + e;
    float v = (k < KREAL) ? W[(size_t)k * NOUT + n] : 0.0f;
    P[idx] = f2bf(v);
}

// ------------------------------------------------------------- WMMA GEMM
// One 16x16 output tile per wave, 4 waves/block. A: bf16 row-major [nRows x K].
// BIAS_RELU=true : out = bf16(relu(acc + bias[n]))              -> OutBf
// BIAS_RELU=false: OutF = acc ; Agg = acc * dinv[m]^2 (self-loop) -> fp32
template <int K, int NOUT, bool BIAS_RELU>
__global__ __launch_bounds__(128) void gemm_wmma(
    const unsigned short* __restrict__ A, const unsigned short* __restrict__ Bp,
    const float* __restrict__ bias, const float* __restrict__ dinv,
    unsigned short* __restrict__ OutBf, float* __restrict__ OutF,
    float* __restrict__ Agg, int nRows) {
    constexpr int KS = K / 32;
    const int wave = threadIdx.x >> 5;
    const int lane = threadIdx.x & 31;
    const int m0 = (blockIdx.x * 4 + wave) * 16;
    if (m0 >= nRows) return;                       // wave-uniform: EXEC stays full

    const int half = lane >> 4;                    // lane-half selects K sub-block
    const int row  = m0 + (lane & 15);
    const unsigned short* arow = A + (size_t)row * K;
    const unsigned short* bp0  = Bp + ((size_t)blockIdx.y * KS * 32 + lane) * 16;

    v8f acc = {0.f, 0.f, 0.f, 0.f, 0.f, 0.f, 0.f, 0.f};
#pragma unroll
    for (int s = 0; s < KS; ++s) {
        Frag256 a, b;
        const unsigned short* ap = arow + s * 32 + half * 16;
        a.q[0] = *(const uint4*)(ap);              // K = k0..k0+7   (this half)
        a.q[1] = *(const uint4*)(ap + 8);          // K = k0+8..+15  (this half)
        const unsigned short* bp = bp0 + (size_t)s * 32 * 16;
        b.q[0] = ((const uint4*)bp)[0];
        b.q[1] = ((const uint4*)bp)[1];
        acc = __builtin_amdgcn_wmma_f32_16x16x32_bf16(
            false, a.v, false, b.v, (short)0, acc, false, false);
    }

    const int n  = blockIdx.y * 16 + (lane & 15);
    const int mb = m0 + half * 8;
    if constexpr (BIAS_RELU) {
        const float bn = bias[n];
#pragma unroll
        for (int j = 0; j < 8; ++j) {
            float v = fmaxf(acc[j] + bn, 0.0f);
            OutBf[(size_t)(mb + j) * NOUT + n] = f2bf(v);
        }
    } else {
#pragma unroll
        for (int j = 0; j < 8; ++j) {
            const int m  = mb + j;
            const float v = acc[j];
            OutF[(size_t)m * NOUT + n] = v;
            const float di = dinv[m];
            Agg[(size_t)m * NOUT + n] = v * di * di;   // fused self-loop term
        }
    }
}

// ------------------------------------------------- edge scatter (one wave/edge)
template <int C>
__global__ __launch_bounds__(256) void edge_scatter(
    const int* __restrict__ src, const int* __restrict__ dst,
    const float* __restrict__ dinv, const float* __restrict__ hw,
    float* __restrict__ agg, int E) {
    constexpr int VPL = C / 32;                    // channels per lane
    int gid  = blockIdx.x * blockDim.x + threadIdx.x;
    int e    = gid >> 5;
    int lane = gid & 31;
    if (e >= E) return;
    const int s = src[e], d = dst[e];
    const float norm = dinv[s] * dinv[d];
    const float* p = hw  + (size_t)s * C + lane * VPL;
    float*       q = agg + (size_t)d * C + lane * VPL;
    float v[VPL];
#pragma unroll
    for (int k = 0; k < VPL; ++k) v[k] = p[k];     // coalesced 128B/wave gather
#pragma unroll
    for (int k = 0; k < VPL; ++k) atomicAdd(q + k, v[k] * norm);
}

// h1b = bf16(relu(agg + bias))  (feeds next WMMA GEMM)
__global__ void finalize_bf16(const float* __restrict__ agg, const float* __restrict__ bias,
                              unsigned short* __restrict__ outb, int nRows, int C) {
    int idx = blockIdx.x * blockDim.x + threadIdx.x;
    if (idx >= nRows * C) return;
    int c = idx % C;
    outb[idx] = f2bf(fmaxf(agg[idx] + bias[c], 0.0f));
}

__global__ void count_kernel(const int* __restrict__ batch, float* __restrict__ cnt, int n) {
    int i = blockIdx.x * blockDim.x + threadIdx.x;
    if (i < n) atomicAdd(&cnt[batch[i]], 1.0f);
}

// relu(agg2 + b2) fused with graph mean-pool accumulation
__global__ void finalize_pool(const float* __restrict__ agg, const float* __restrict__ bias,
                              const int* __restrict__ batch, float* __restrict__ pool, int nRows) {
    int idx = blockIdx.x * blockDim.x + threadIdx.x;
    if (idx >= nRows * LATENT) return;
    int i = idx >> 6, c = idx & 63;
    float v = fmaxf(agg[idx] + bias[c], 0.0f);
    atomicAdd(&pool[(size_t)batch[i] * LATENT + c], v);
}

__global__ void mean_kernel(const float* __restrict__ pool, const float* __restrict__ cnt,
                            float* __restrict__ out, int G) {
    int idx = blockIdx.x * blockDim.x + threadIdx.x;
    if (idx >= G * LATENT) return;
    int g = idx >> 6;
    out[idx] = pool[idx] / fmaxf(cnt[g], 1.0f);
}

// ---------------------------------------------------------------------------
extern "C" void kernel_launch(void* const* d_in, const int* in_sizes, int n_in,
                              void* d_out, int out_size, void* d_ws, size_t ws_size,
                              hipStream_t stream) {
    const float* x     = (const float*)d_in[0];
    const float* W_emb = (const float*)d_in[1];
    const float* b_emb = (const float*)d_in[2];
    const float* W1    = (const float*)d_in[3];
    const float* b1    = (const float*)d_in[4];
    const float* W2    = (const float*)d_in[5];
    const float* b2    = (const float*)d_in[6];
    const int*   ei    = (const int*)d_in[7];     // [2 x E]: src then dst
    const int*   batch = (const int*)d_in[8];

    const int N = in_sizes[0] / NODE_DIM;
    const int E = in_sizes[7] / 2;
    const int G = out_size / LATENT;
    const int* src = ei;
    const int* dst = ei + E;

    // ---- workspace carve-out (256B aligned); hot regions aliased where dead
    char* ws = (char*)d_ws;
    size_t off = 0;
    auto take = [&](size_t bytes) -> void* {
        void* p = ws + off;
        off += (bytes + 255) & ~(size_t)255;
        return p;
    };
    float*          dinv = (float*)take((size_t)N * 4);
    unsigned short* xb   = (unsigned short*)take((size_t)N * KPAD_EMB * 2);
    unsigned short* h0b  = (unsigned short*)take((size_t)N * HIDDEN * 2);   // reused as h1b
    float*          hw1  = (float*)take((size_t)N * HIDDEN * 4);            // reused as hw2+agg2
    float*          agg1 = (float*)take((size_t)N * HIDDEN * 4);
    float*          pool = (float*)take((size_t)G * LATENT * 4);
    float*          cnt  = (float*)take((size_t)G * 4);
    unsigned short* Wpe  = (unsigned short*)take((size_t)(HIDDEN/16)*(KPAD_EMB/32)*512*2);
    unsigned short* Wp1  = (unsigned short*)take((size_t)(HIDDEN/16)*(HIDDEN/32)*512*2);
    unsigned short* Wp2  = (unsigned short*)take((size_t)(LATENT/16)*(HIDDEN/32)*512*2);
    unsigned short* h1b  = h0b;                       // alias (h0b dead after GEMM1)
    float*          hw2  = hw1;                       // alias (hw1 dead after scatter1)
    float*          agg2 = hw1 + (size_t)N * LATENT;  // second half of hw1 region

    auto blocks = [](long long n, int b) { return (unsigned)((n + b - 1) / b); };

    // ---- degree / normalization
    fill_f32<<<blocks(N, 256), 256, 0, stream>>>(dinv, 1.0f, N);            // self-loop
    deg_kernel<<<blocks(E, 256), 256, 0, stream>>>(dst, dinv, E);
    rsqrt_kernel<<<blocks(N, 256), 256, 0, stream>>>(dinv, N);
    fill_f32<<<blocks(G * LATENT, 256), 256, 0, stream>>>(pool, 0.0f, G * LATENT);
    fill_f32<<<blocks(G, 256), 256, 0, stream>>>(cnt, 0.0f, G);
    count_kernel<<<blocks(N, 256), 256, 0, stream>>>(batch, cnt, N);

    // ---- bf16 staging: activations + weight fragments
    pad_x_bf16<<<blocks((long long)N * KPAD_EMB, 256), 256, 0, stream>>>(x, xb, N);
    pack_w<NODE_DIM, KPAD_EMB, HIDDEN><<<blocks((HIDDEN/16)*(KPAD_EMB/32)*512, 256), 256, 0, stream>>>(W_emb, Wpe);
    pack_w<HIDDEN,   HIDDEN,   HIDDEN><<<blocks((HIDDEN/16)*(HIDDEN/32)*512, 256), 256, 0, stream>>>(W1, Wp1);
    pack_w<HIDDEN,   HIDDEN,   LATENT><<<blocks((LATENT/16)*(HIDDEN/32)*512, 256), 256, 0, stream>>>(W2, Wp2);

    const unsigned mtb = blocks((N + 15) / 16, 4);    // row-tile blocks (4 waves/block)

    // ---- layer 0: h0 = relu(x @ W_emb + b_emb)  -> bf16
    gemm_wmma<KPAD_EMB, HIDDEN, true><<<dim3(mtb, HIDDEN / 16), 128, 0, stream>>>(
        xb, Wpe, b_emb, nullptr, h0b, nullptr, nullptr, N);

    // ---- conv1: hw1 = h0 @ W1 ; agg1 = hw1*dinv^2 ; scatter ; relu(+b1) -> bf16
    gemm_wmma<HIDDEN, HIDDEN, false><<<dim3(mtb, HIDDEN / 16), 128, 0, stream>>>(
        h0b, Wp1, nullptr, dinv, nullptr, hw1, agg1, N);
    edge_scatter<HIDDEN><<<blocks((long long)E * 32, 256), 256, 0, stream>>>(
        src, dst, dinv, hw1, agg1, E);
    finalize_bf16<<<blocks((long long)N * HIDDEN, 256), 256, 0, stream>>>(
        agg1, b1, h1b, N, HIDDEN);

    // ---- conv2: hw2 = h1 @ W2 ; agg2 ; scatter ; relu(+b2) fused with pooling
    gemm_wmma<HIDDEN, LATENT, false><<<dim3(mtb, LATENT / 16), 128, 0, stream>>>(
        h1b, Wp2, nullptr, dinv, nullptr, hw2, agg2, N);
    edge_scatter<LATENT><<<blocks((long long)E * 32, 256), 256, 0, stream>>>(
        src, dst, dinv, hw2, agg2, E);
    finalize_pool<<<blocks((long long)N * LATENT, 256), 256, 0, stream>>>(
        agg2, b2, batch, pool, N);

    // ---- global mean pool -> output [G x 64]
    mean_kernel<<<blocks(G * LATENT, 256), 256, 0, stream>>>(pool, cnt, (float*)d_out, G);
    (void)n_in; (void)ws_size;
}
